// Decoder_59811714564230
// MI455X (gfx1250) — matrix operand order repaired and checked by reading, not compiled, not run on hardware
//
#include <hip/hip_runtime.h>
#include <hip/hip_bf16.h>

// ---------------------------------------------------------------------------
// CDNA5 (gfx1250, wave32) WMMA types
// ---------------------------------------------------------------------------
typedef __attribute__((ext_vector_type(16))) __bf16       v16bf;
typedef __attribute__((ext_vector_type(8)))  float        v8f;
typedef __attribute__((ext_vector_type(4)))  unsigned int u32x4;

__device__ __forceinline__ __bf16 f2bf(float f) {
  unsigned u = __builtin_bit_cast(unsigned, f);
  u += 0x7FFFu + ((u >> 16) & 1u);                 // round-to-nearest-even
  unsigned short h = (unsigned short)(u >> 16);
  return __builtin_bit_cast(__bf16, h);
}
__device__ __forceinline__ float sigf(float x) { return 1.f / (1.f + __expf(-x)); }

// A-fragment (16x32 bf16): lane l<16 -> row l, K chunks {0..7},{16..23};
//                          lane l>=16 -> row l-16, K chunks {8..15},{24..31}.
// B-fragment (32x16 bf16): lane holds column N=lane&15; lanes 0-15 K chunks {0..7},{16..23},
//                          lanes 16-31 K chunks {8..15},{24..31} (same as A).
union frag { u32x4 q[2]; v16bf v; };

__device__ __forceinline__ void load_a(frag& A, const __bf16* row, int k0, int half) {
  const __bf16* ap = row + k0 + half * 8;
  A.q[0] = *(const u32x4*)(ap);
  A.q[1] = *(const u32x4*)(ap + 16);
}
__device__ __forceinline__ void load_b(frag& B, const __bf16* row, int k0, int half) {
  const __bf16* bp = row + k0 + half * 8;
  B.q[0] = *(const u32x4*)(bp);
  B.q[1] = *(const u32x4*)(bp + 16);
}
// C/D layout: lane holds column n = lane&15; VGPR v -> row v + 8*(lane>>4).
__device__ __forceinline__ void store_tile(const v8f& acc, float* __restrict__ out,
                                           int ldo, int mbase, int ncol, float bv,
                                           const float* __restrict__ rowmask,
                                           int flags, int half) {
#pragma unroll
  for (int v = 0; v < 8; ++v) {
    int    mrow = mbase + v + 8 * half;
    size_t idx  = (size_t)mrow * ldo + ncol;
    float  r    = acc[v] + bv;
    if (flags & 2) r += out[idx];
    if (flags & 1) r = fmaxf(r, 0.f);
    if (rowmask)   r *= rowmask[mrow];
    out[idx] = r;
  }
}

// ---------------------------------------------------------------------------
// 16x16-per-wave GEMM (max parallelism; used for the small sequential GEMMs)
// out[M,N](f32) = X[M,K](bf16) * W[N,K]^T ; flags: 1=relu, 2=accumulate
// ---------------------------------------------------------------------------
__global__ __launch_bounds__(256) void k_gemm(
    const __bf16* __restrict__ X, const __bf16* __restrict__ W,
    const float* __restrict__ bias, const float* __restrict__ rowmask,
    float* __restrict__ out, int N, int K, int ldx, int ldw, int ldo, int flags)
{
  int lane = threadIdx.x & 31;
  int wave = threadIdx.x >> 5;
  int tN   = blockIdx.x * 8 + wave;
  if (tN * 16 >= N) return;
  int tM   = blockIdx.y;
  int l15  = lane & 15;
  int half = lane >> 4;

  const __bf16* aRow = X + (size_t)(tM * 16 + l15) * ldx;
  const __bf16* bRow = W + (size_t)(tN * 16 + l15) * ldw;

  v8f acc = {};
  for (int k0 = 0; k0 < K; k0 += 32) {
    frag A, B;
    load_a(A, aRow, k0, half);
    load_b(B, bRow, k0, half);
    __builtin_prefetch(bRow + k0 + half * 16 + 64, 0, 1);
    acc = __builtin_amdgcn_wmma_f32_16x16x32_bf16(
        false, A.v, false, B.v, (short)0, acc, false, false);
  }
  int   ncol = tN * 16 + l15;
  float bv   = bias ? bias[ncol] : 0.f;
  store_tile(acc, out, ldo, tM * 16, ncol, bv, rowmask, flags, half);
}

// ---------------------------------------------------------------------------
// 32x32-per-wave register-blocked GEMM (2M x 2N, 4 accumulators): each A/B
// fragment feeds two wmmas -> 1KB loaded per wmma instead of 2KB. Used for the
// large GEMMs (V, WV, vocab projection). M, N must be multiples of 32.
// ---------------------------------------------------------------------------
__global__ __launch_bounds__(256) void k_gemm2(
    const __bf16* __restrict__ X, const __bf16* __restrict__ W,
    const float* __restrict__ bias, const float* __restrict__ rowmask,
    float* __restrict__ out, int N, int K, int ldx, int ldw, int ldo, int flags)
{
  int lane = threadIdx.x & 31;
  int wave = threadIdx.x >> 5;
  int tN   = blockIdx.x * 8 + wave;          // 32-column tile index
  if (tN * 32 >= N) return;                  // guard (N=20000 -> 625 tiles exact)
  int tM   = blockIdx.y;                     // 32-row tile index
  int l15  = lane & 15;
  int half = lane >> 4;

  const __bf16* aRow0 = X + (size_t)(tM * 32 + l15) * ldx;
  const __bf16* aRow1 = aRow0 + (size_t)16 * ldx;
  const __bf16* bRow0 = W + (size_t)(tN * 32 + l15) * ldw;
  const __bf16* bRow1 = bRow0 + (size_t)16 * ldw;

  v8f a00 = {}, a01 = {}, a10 = {}, a11 = {};
  for (int k0 = 0; k0 < K; k0 += 32) {
    frag A0, A1, B0, B1;
    load_a(A0, aRow0, k0, half);
    load_a(A1, aRow1, k0, half);
    load_b(B0, bRow0, k0, half);
    load_b(B1, bRow1, k0, half);
    __builtin_prefetch(bRow0 + k0 + half * 16 + 64, 0, 1);
    __builtin_prefetch(bRow1 + k0 + half * 16 + 64, 0, 1);
    a00 = __builtin_amdgcn_wmma_f32_16x16x32_bf16(false, A0.v, false, B0.v, (short)0, a00, false, false);
    a01 = __builtin_amdgcn_wmma_f32_16x16x32_bf16(false, A0.v, false, B1.v, (short)0, a01, false, false);
    a10 = __builtin_amdgcn_wmma_f32_16x16x32_bf16(false, A1.v, false, B0.v, (short)0, a10, false, false);
    a11 = __builtin_amdgcn_wmma_f32_16x16x32_bf16(false, A1.v, false, B1.v, (short)0, a11, false, false);
  }
  int   nc0 = tN * 32 + l15, nc1 = nc0 + 16;
  float bv0 = bias ? bias[nc0] : 0.f;
  float bv1 = bias ? bias[nc1] : 0.f;
  int   m0  = tM * 32, m1 = m0 + 16;
  store_tile(a00, out, ldo, m0, nc0, bv0, rowmask, flags, half);
  store_tile(a01, out, ldo, m0, nc1, bv1, rowmask, flags, half);
  store_tile(a10, out, ldo, m1, nc0, bv0, rowmask, flags, half);
  store_tile(a11, out, ldo, m1, nc1, bv1, rowmask, flags, half);
}

// ---------------------------------------------------------------------------
// Small helper kernels
// ---------------------------------------------------------------------------
__global__ void k_sort(const int* __restrict__ capl, int* __restrict__ sind,
                       int* __restrict__ dlen)
{
  __shared__ int keys[128];
  int tid = threadIdx.x;
  keys[tid] = capl[tid];
  __syncthreads();
  int k = keys[tid], pos = 0;
  for (int j = 0; j < 128; ++j) {
    int kj = keys[j];
    pos += (kj > k) || (kj == k && j < tid);   // stable argsort(-cl)
  }
  sind[pos] = tid;
  dlen[pos] = k - 1;
}

__global__ void k_tail(const int* __restrict__ caps, const int* __restrict__ sind,
                       const int* __restrict__ dlen, float* __restrict__ o)
{
  int i = blockIdx.x * blockDim.x + threadIdx.x;
  if (i < 2560)      { int r = i / 20, t = i % 20; o[i] = (float)caps[sind[r] * 20 + t]; }
  else if (i < 2688) { o[i] = (float)dlen[i - 2560]; }
  else if (i < 2816) { o[i] = (float)sind[i - 2688]; }
}

__global__ void k_cvt(const float* __restrict__ s, __bf16* __restrict__ d, int n) {
  int i = blockIdx.x * blockDim.x + threadIdx.x;
  if (i < n) d[i] = f2bf(s[i]);
}

__global__ void k_addvec(const float* __restrict__ a, const float* __restrict__ b,
                         float* __restrict__ o, int n) {
  int i = blockIdx.x * blockDim.x + threadIdx.x;
  if (i < n) o[i] = a[i] + b[i];
}

__global__ void k_gather(const float* __restrict__ src, __bf16* __restrict__ dst,
                         const int* __restrict__ sind, int chunk, int n) {
  int i = blockIdx.x * blockDim.x + threadIdx.x;
  if (i < n) {
    int b = i / chunk, off = i - b * chunk;
    dst[i] = f2bf(src[(size_t)sind[b] * chunk + off]);
  }
}

// inputX (bf16, 128x20x1024): [:512]=emb_W[caps_s], [512:]=vg[b]
__global__ void k_buildX(const float* __restrict__ embW, const int* __restrict__ caps,
                         const int* __restrict__ sind, const float* __restrict__ vg,
                         __bf16* __restrict__ dst)
{
  int i = blockIdx.x * blockDim.x + threadIdx.x;
  if (i >= 128 * 20 * 1024) return;
  int b = i / 20480, r = i % 20480, t = r / 1024, col = r & 1023;
  float v;
  if (col < 512) { int c = caps[sind[b] * 20 + t]; v = embW[(size_t)c * 512 + col]; }
  else           { v = vg[b * 512 + (col - 512)]; }
  dst[i] = f2bf(v);
}

__global__ void k_zero(float* __restrict__ h, __bf16* __restrict__ hb,
                       float* __restrict__ m) {
  int i = blockIdx.x * blockDim.x + threadIdx.x;
  if (i < 128 * 512) { h[i] = 0.f; hb[i] = f2bf(0.f); m[i] = 0.f; }
}

// LSTM cell + sentinel gate (gates order i,f,g,o per jnp.split)
__global__ void k_cell(const float* __restrict__ gates, const float* __restrict__ gpre,
                       float* __restrict__ h, __bf16* __restrict__ hb, float* __restrict__ m,
                       float* __restrict__ hnf, __bf16* __restrict__ hnb,
                       float* __restrict__ sf,  __bf16* __restrict__ sb,
                       const int* __restrict__ dlen, int t)
{
  int idx = blockIdx.x * blockDim.x + threadIdx.x;
  if (idx >= 128 * 512) return;
  int b = idx >> 9, n = idx & 511;
  const float* g = gates + (size_t)b * 2048;
  float i_ = sigf(g[n]), f_ = sigf(g[n + 512]);
  float gg = tanhf(g[n + 1024]), o_ = sigf(g[n + 1536]);
  float mn = f_ * m[idx] + i_ * gg;
  float hn = o_ * tanhf(mn);
  float gt = sigf(gpre[idx]);
  float sv = gt * tanhf(mn);
  hnf[idx] = hn; hnb[idx] = f2bf(hn);
  sf[idx]  = sv; sb[idx]  = f2bf(sv);
  bool act = t < dlen[b];
  float hc = act ? hn : h[idx];
  float mc = act ? mn : m[idx];
  h[idx] = hc; hb[idx] = f2bf(hc); m[idx] = mc;
}

// z[b,p] = sum_a tanh(WV[b,p,a] + Wh[b,a]) * wa[a] + ba   (one wave per (b,p))
__global__ __launch_bounds__(256) void k_attn1(
    const float* __restrict__ WV, const float* __restrict__ Wh,
    const float* __restrict__ wa, const float* __restrict__ ba,
    float* __restrict__ z)
{
  int lane = threadIdx.x & 31, wave = threadIdx.x >> 5;
  int gid = blockIdx.x * 8 + wave;
  if (gid >= 128 * 49) return;
  int b = gid / 49;
  const float* wv = WV + (size_t)gid * 512;
  const float* wh = Wh + (size_t)b * 512;
  float s = 0.f;
#pragma unroll
  for (int j = 0; j < 16; ++j) {
    int a = lane + j * 32;
    s += tanhf(wv[a] + wh[a]) * wa[a];
  }
  for (int m = 16; m >= 1; m >>= 1) s += __shfl_xor(s, m, 32);
  if (lane == 0) z[gid] = s + ba[0];
}

// Per-b: zhat, softmax(z)->alpha, softmax([z,zhat])[-1]->beta, c, q=c_hat+h_new
__global__ __launch_bounds__(256) void k_attn2(
    const float* __restrict__ z, const float* __restrict__ Ws, const float* __restrict__ Wh,
    const float* __restrict__ wahat, const float* __restrict__ bahat,
    const float* __restrict__ V, const float* __restrict__ s, const float* __restrict__ hn,
    const int* __restrict__ dlen, __bf16* __restrict__ Qb, float* __restrict__ actf, int t)
{
  int b = blockIdx.x, tid = threadIdx.x;
  __shared__ float red[256];
  __shared__ float alpha[49];
  __shared__ float betaS;
  float p0 = 0.f;
  for (int a = tid; a < 512; a += 256)
    p0 += tanhf(Ws[b * 512 + a] + Wh[b * 512 + a]) * wahat[a];
  red[tid] = p0; __syncthreads();
  for (int off = 128; off > 0; off >>= 1) {
    if (tid < off) red[tid] += red[tid + off];
    __syncthreads();
  }
  if (tid == 0) {
    float zhat = red[0] + bahat[0];
    float zm = -1e30f;
    for (int p = 0; p < 49; ++p) zm = fmaxf(zm, z[b * 49 + p]);
    float se = 0.f;
    for (int p = 0; p < 49; ++p) { float e = __expf(z[b * 49 + p] - zm); alpha[p] = e; se += e; }
    float inv = 1.f / se;
    for (int p = 0; p < 49; ++p) alpha[p] *= inv;
    float m2 = fmaxf(zm, zhat);
    float d2 = se * __expf(zm - m2) + __expf(zhat - m2);
    betaS = __expf(zhat - m2) / d2;
    actf[b * 19 + t] = (t < dlen[b]) ? 1.f : 0.f;
  }
  __syncthreads();
  float beta = betaS;
  for (int e = tid; e < 512; e += 256) {
    float c = 0.f;
    for (int p = 0; p < 49; ++p) c += alpha[p] * V[((size_t)b * 49 + p) * 512 + e];
    float ch = beta * s[b * 512 + e] + (1.f - beta) * c;
    Qb[((size_t)b * 19 + t) * 512 + e] = f2bf(ch + hn[b * 512 + e]);
  }
}

// ---------------------------------------------------------------------------
// Host orchestration
// ---------------------------------------------------------------------------
extern "C" void kernel_launch(void* const* d_in, const int* in_sizes, int n_in,
                              void* d_out, int out_size, void* d_ws, size_t ws_size,
                              hipStream_t stream)
{
  const float* enc  = (const float*)d_in[0];
  const float* gimg = (const float*)d_in[1];
  const int*   caps = (const int*)d_in[2];
  const int*   capl = (const int*)d_in[3];
  const float* embW = (const float*)d_in[4];
  const float* W_ih = (const float*)d_in[5];
  const float* W_hh = (const float*)d_in[6];
  const float* b_ih = (const float*)d_in[7];
  const float* b_hh = (const float*)d_in[8];
  const float* Wg   = (const float*)d_in[9];
  const float* bg   = (const float*)d_in[10];
  const float* Wi   = (const float*)d_in[11];
  const float* bi   = (const float*)d_in[12];
  const float* Wx   = (const float*)d_in[13];
  const float* bx   = (const float*)d_in[14];
  const float* Wph  = (const float*)d_in[15];
  const float* bph  = (const float*)d_in[16];
  const float* aWh  = (const float*)d_in[17];
  const float* abh  = (const float*)d_in[18];
  const float* aWs  = (const float*)d_in[19];
  const float* abs_ = (const float*)d_in[20];
  const float* aWV  = (const float*)d_in[21];
  const float* abV  = (const float*)d_in[22];
  const float* awa  = (const float*)d_in[23];
  const float* aba  = (const float*)d_in[24];
  const float* awah = (const float*)d_in[25];
  const float* abah = (const float*)d_in[26];
  const float* fW   = (const float*)d_in[27];
  const float* fb   = (const float*)d_in[28];
  float* out = (float*)d_out;

  const size_t PRED = (size_t)128 * 19 * 20000;

  // workspace carve-out (256B aligned)
  char* wp = (char*)d_ws;
  auto alloc = [&](size_t bytes) -> void* {
    void* p = (void*)wp; wp += (bytes + 255) & ~(size_t)255; return p;
  };
  int*    sind  = (int*)alloc(128 * 4);
  int*    dlen  = (int*)alloc(128 * 4);
  __bf16* encb  = (__bf16*)alloc((size_t)6272 * 2048 * 2);
  __bf16* gimgb = (__bf16*)alloc((size_t)128 * 2048 * 2);
  __bf16* Wib   = (__bf16*)alloc((size_t)512 * 2048 * 2);
  __bf16* Wgb   = (__bf16*)alloc((size_t)512 * 2048 * 2);
  __bf16* Wihb  = (__bf16*)alloc((size_t)2048 * 1024 * 2);
  __bf16* Whhb  = (__bf16*)alloc((size_t)2048 * 512 * 2);
  __bf16* Wxb   = (__bf16*)alloc((size_t)512 * 1024 * 2);
  __bf16* Wphb  = (__bf16*)alloc((size_t)512 * 512 * 2);
  __bf16* aWhb  = (__bf16*)alloc((size_t)512 * 512 * 2);
  __bf16* aWsb  = (__bf16*)alloc((size_t)512 * 512 * 2);
  __bf16* aWVb  = (__bf16*)alloc((size_t)512 * 512 * 2);
  __bf16* fWb   = (__bf16*)alloc((size_t)20000 * 512 * 2);
  float*  bsum  = (float*)alloc(2048 * 4);
  float*  bxs   = (float*)alloc(512 * 4);
  float*  vg    = (float*)alloc((size_t)128 * 512 * 4);
  __bf16* inXb  = (__bf16*)alloc((size_t)128 * 20 * 1024 * 2);
  float*  Vf    = (float*)alloc((size_t)6272 * 512 * 4);
  __bf16* Vb    = (__bf16*)alloc((size_t)6272 * 512 * 2);
  float*  WVf   = (float*)alloc((size_t)6272 * 512 * 4);
  float*  h     = (float*)alloc((size_t)128 * 512 * 4);
  __bf16* hb    = (__bf16*)alloc((size_t)128 * 512 * 2);
  float*  m     = (float*)alloc((size_t)128 * 512 * 4);
  float*  hnf   = (float*)alloc((size_t)128 * 512 * 4);
  __bf16* hnb   = (__bf16*)alloc((size_t)128 * 512 * 2);
  float*  sf    = (float*)alloc((size_t)128 * 512 * 4);
  __bf16* sb    = (__bf16*)alloc((size_t)128 * 512 * 2);
  float*  gates = (float*)alloc((size_t)128 * 2048 * 4);
  float*  gpre  = (float*)alloc((size_t)128 * 512 * 4);
  float*  Whf   = (float*)alloc((size_t)128 * 512 * 4);
  float*  Wsf   = (float*)alloc((size_t)128 * 512 * 4);
  float*  zf    = (float*)alloc((size_t)128 * 49 * 4);
  __bf16* Qb    = (__bf16*)alloc((size_t)2432 * 512 * 2);
  float*  actf  = (float*)alloc((size_t)2432 * 4);

  auto cvt = [&](const float* s, __bf16* d, int n) {
    k_cvt<<<dim3((n + 255) / 256), 256, 0, stream>>>(s, d, n);
  };
  // 16x16 tiles: used for the small latency-bound per-step GEMMs
  auto gemm = [&](const __bf16* X, const __bf16* W, const float* bias, const float* rm,
                  float* o, int M, int N, int K, int ldx, int ldw, int ldo, int flags) {
    dim3 g((N / 16 + 7) / 8, M / 16);
    k_gemm<<<g, 256, 0, stream>>>(X, W, bias, rm, o, N, K, ldx, ldw, ldo, flags);
  };
  // 32x32 register-blocked tiles: used for the large throughput GEMMs
  auto gemm2 = [&](const __bf16* X, const __bf16* W, const float* bias, const float* rm,
                   float* o, int M, int N, int K, int ldx, int ldw, int ldo, int flags) {
    dim3 g((N / 32 + 7) / 8, M / 32);
    k_gemm2<<<g, 256, 0, stream>>>(X, W, bias, rm, o, N, K, ldx, ldw, ldo, flags);
  };

  // ---- sort + int outputs (written as float values) -----------------------
  k_sort<<<1, 128, 0, stream>>>(capl, sind, dlen);
  k_tail<<<11, 256, 0, stream>>>(caps, sind, dlen, out + PRED);

  // ---- per-call bf16 weight conversions -----------------------------------
  cvt(Wi,  Wib,  512 * 2048);
  cvt(Wg,  Wgb,  512 * 2048);
  cvt(W_ih, Wihb, 2048 * 1024);
  cvt(W_hh, Whhb, 2048 * 512);
  cvt(Wx,  Wxb,  512 * 1024);
  cvt(Wph, Wphb, 512 * 512);
  cvt(aWh, aWhb, 512 * 512);
  cvt(aWs, aWsb, 512 * 512);
  cvt(aWV, aWVb, 512 * 512);
  cvt(fW,  fWb,  20000 * 512);
  k_addvec<<<8, 256, 0, stream>>>(b_ih, b_hh, bsum, 2048);
  k_addvec<<<2, 256, 0, stream>>>(bx, bph, bxs, 512);

  // ---- sorted gathers -----------------------------------------------------
  {
    int n = 128 * 49 * 2048;
    k_gather<<<(n + 255) / 256, 256, 0, stream>>>(enc, encb, sind, 49 * 2048, n);
    n = 128 * 2048;
    k_gather<<<(n + 255) / 256, 256, 0, stream>>>(gimg, gimgb, sind, 2048, n);
  }

  // ---- image-side GEMMs ---------------------------------------------------
  gemm (gimgb, Wgb, bg, nullptr, vg, 128, 512, 2048, 2048, 2048, 512, 1);     // vg (relu)
  gemm2(encb,  Wib, bi, nullptr, Vf, 6272, 512, 2048, 2048, 2048, 512, 1);    // V  (relu)
  cvt(Vf, Vb, 6272 * 512);
  k_buildX<<<(128 * 20 * 1024 + 255) / 256, 256, 0, stream>>>(embW, caps, sind, vg, inXb);
  gemm2(Vb, aWVb, abV, nullptr, WVf, 6272, 512, 512, 512, 512, 512, 0);       // WV

  // ---- LSTM scan (small GEMMs + elementwise) ------------------------------
  k_zero<<<(128 * 512 + 255) / 256, 256, 0, stream>>>(h, hb, m);
  for (int t = 0; t < 19; ++t) {
    const __bf16* xt = inXb + (size_t)t * 1024;   // row stride 20480 per batch item
    gemm(xt, Wihb, bsum, nullptr, gates, 128, 2048, 1024, 20480, 1024, 2048, 0);
    gemm(hb, Whhb, nullptr, nullptr, gates, 128, 2048, 512, 512, 512, 2048, 2);
    gemm(xt, Wxb,  bxs, nullptr, gpre, 128, 512, 1024, 20480, 1024, 512, 0);
    gemm(hb, Wphb, nullptr, nullptr, gpre, 128, 512, 512, 512, 512, 512, 2);
    k_cell<<<(128 * 512 + 255) / 256, 256, 0, stream>>>(gates, gpre, h, hb, m,
                                                        hnf, hnb, sf, sb, dlen, t);
    gemm(hnb, aWhb, abh,  nullptr, Whf, 128, 512, 512, 512, 512, 512, 0);
    gemm(sb,  aWsb, abs_, nullptr, Wsf, 128, 512, 512, 512, 512, 512, 0);
    k_attn1<<<784, 256, 0, stream>>>(WVf, Whf, awa, aba, zf);
    k_attn2<<<128, 256, 0, stream>>>(zf, Wsf, Whf, awah, abah, Vf, sf, hnf,
                                     dlen, Qb, actf, t);
  }

  // ---- hoisted vocab projection: (2432 x 20000 x 512), masked rows -> 0 ----
  gemm2(Qb, fWb, fb, actf, out, 2432, 20000, 512, 512, 512, 20000, 0);
}
